// AdvancedGNN_12790412607663
// MI455X (gfx1250) — matrix operand order repaired and checked by reading, not compiled, not run on hardware
//
#include <hip/hip_runtime.h>

#define F_IN  128
#define HEADS 4
#define HID   64
#define CH    (HEADS*HID)   /* 256 */
#define NGR   500
#define OUTD  128
#define NCLS  2
#define SLOPE 0.2f
#define BNEPS 1e-5f

typedef __attribute__((ext_vector_type(16))) __bf16 v16bf;
typedef __attribute__((ext_vector_type(8)))  __bf16 v8bf;
typedef __attribute__((ext_vector_type(8)))  float  v8f;

static __device__ __forceinline__ unsigned fenc(float f) {
    unsigned u = __float_as_uint(f);
    return (u & 0x80000000u) ? ~u : (u | 0x80000000u);
}
static __device__ __forceinline__ float fdec(unsigned u) {
    unsigned v = (u & 0x80000000u) ? (u & 0x7FFFFFFFu) : ~u;
    return __uint_as_float(v);
}
static __device__ __forceinline__ float lrelu(float v) {
    return v > 0.0f ? v : SLOPE * v;
}

// ---------------------------------------------------------------------------
// f32 -> bf16 flat convert
__global__ void k_cvt(const float* __restrict__ in, __bf16* __restrict__ out,
                      size_t n) {
    size_t i = (size_t)blockIdx.x * blockDim.x + threadIdx.x;
    if (i < n) out[i] = (__bf16)in[i];
}

// W [K, Nout] f32 row-major  ->  Wt [Nout, K] bf16 row-major
__global__ void k_wt(const float* __restrict__ W, __bf16* __restrict__ Wt,
                     int K, int Nout) {
    int i = blockIdx.x * blockDim.x + threadIdx.x;
    if (i >= K * Nout) return;
    int n = i / K, k = i % K;
    Wt[i] = (__bf16)W[(size_t)k * Nout + n];
}

// ---------------------------------------------------------------------------
// Wave-level 32x32 macro-tile GEMM: C[M,Nout] = A[M,K](bf16) * Bt[Nout,K]^T.
// Each wave owns a 32x32 output tile = 2x2 grid of 16x16 WMMA tiles:
// 2 A fragments + 2 B fragments -> 4 V_WMMA_F32_16X16X32_BF16 per K-step.
// Fragment layouts per ISA 7.12.2 (16-bit A and B: half-lanes hold K=h*8+{0..7}
// and +{16..23}).
template <bool BIAS, bool RELU>
__global__ __launch_bounds__(256) void k_gemm_bf16(
    const __bf16* __restrict__ A, const __bf16* __restrict__ Bt,
    const float* __restrict__ bias, float* __restrict__ C,
    int M, int K, int Nout) {
    const int lane = threadIdx.x & 31;
    const int wave = threadIdx.x >> 5;
    const int tilesN = Nout >> 5;        // 32-wide column tiles
    const int tilesM = (M + 31) >> 5;    // 32-tall row tiles (clamped)
    int tile = blockIdx.x * 8 + wave;
    if (tile >= tilesM * tilesN) return;
    const int tm = tile / tilesN, tn = tile % tilesN;
    const int row0 = tm << 5, col0 = tn << 5;
    const int half = lane >> 4, l16 = lane & 15;

    int ar0 = row0 + l16;      if (ar0 >= M) ar0 = M - 1;  // clamp; masked at store
    int ar1 = row0 + 16 + l16; if (ar1 >= M) ar1 = M - 1;
    const __bf16* ap0 = A + (size_t)ar0 * K + half * 8;
    const __bf16* ap1 = A + (size_t)ar1 * K + half * 8;
    const __bf16* bp0 = Bt + (size_t)(col0 + l16) * K + half * 8;
    const __bf16* bp1 = Bt + (size_t)(col0 + 16 + l16) * K + half * 8;

    v8f acc00 = {0.f,0.f,0.f,0.f,0.f,0.f,0.f,0.f};
    v8f acc01 = acc00, acc10 = acc00, acc11 = acc00;

    for (int kk = 0; kk < K; kk += 32) {
        __builtin_prefetch(ap0 + kk + 128, 0, 1);
        __builtin_prefetch(bp0 + kk + 128, 0, 1);
        union { v16bf v; v8bf h[2]; } a0, a1, b0, b1;
        a0.h[0] = *(const v8bf*)(ap0 + kk);
        a0.h[1] = *(const v8bf*)(ap0 + kk + 16);
        a1.h[0] = *(const v8bf*)(ap1 + kk);
        a1.h[1] = *(const v8bf*)(ap1 + kk + 16);
        b0.h[0] = *(const v8bf*)(bp0 + kk);
        b0.h[1] = *(const v8bf*)(bp0 + kk + 16);
        b1.h[0] = *(const v8bf*)(bp1 + kk);
        b1.h[1] = *(const v8bf*)(bp1 + kk + 16);
        acc00 = __builtin_amdgcn_wmma_f32_16x16x32_bf16(
            false, a0.v, false, b0.v, (short)0, acc00, false, false);
        acc01 = __builtin_amdgcn_wmma_f32_16x16x32_bf16(
            false, a0.v, false, b1.v, (short)0, acc01, false, false);
        acc10 = __builtin_amdgcn_wmma_f32_16x16x32_bf16(
            false, a1.v, false, b0.v, (short)0, acc10, false, false);
        acc11 = __builtin_amdgcn_wmma_f32_16x16x32_bf16(
            false, a1.v, false, b1.v, (short)0, acc11, false, false);
    }

    // C layout per 16x16 tile: VGPR r -> M = r + 8*half, N = l16
    const int c0 = col0 + l16, c1 = col0 + 16 + l16;
    float bia0 = BIAS ? bias[c0] : 0.0f;
    float bia1 = BIAS ? bias[c1] : 0.0f;
#pragma unroll
    for (int r = 0; r < 8; r++) {
        int rowa = row0 + r + half * 8;
        int rowb = rowa + 16;
        if (rowa < M) {
            float v0 = acc00[r] + bia0, v1 = acc01[r] + bia1;
            if (RELU) { v0 = fmaxf(v0, 0.f); v1 = fmaxf(v1, 0.f); }
            C[(size_t)rowa * Nout + c0] = v0;
            C[(size_t)rowa * Nout + c1] = v1;
        }
        if (rowb < M) {
            float v0 = acc10[r] + bia0, v1 = acc11[r] + bia1;
            if (RELU) { v0 = fmaxf(v0, 0.f); v1 = fmaxf(v1, 0.f); }
            C[(size_t)rowb * Nout + c0] = v0;
            C[(size_t)rowb * Nout + c1] = v1;
        }
    }
}

// ---------------------------------------------------------------------------
// a_src[n,h] = <H[n,h,:], att_src[h,:]> ; a_dst likewise
__global__ void k_att(const float* __restrict__ H, const float* __restrict__ atts,
                      const float* __restrict__ attd, float* __restrict__ as_,
                      float* __restrict__ ad_, int N) {
    int i = blockIdx.x * blockDim.x + threadIdx.x;
    if (i >= N * HEADS) return;
    int n = i >> 2, h = i & 3;
    const float* hp = H + (size_t)n * CH + h * HID;
    const float* sp = atts + h * HID;
    const float* dp = attd + h * HID;
    float s = 0.f, d = 0.f;
    for (int c = 0; c < HID; c++) { float v = hp[c]; s += v * sp[c]; d += v * dp[c]; }
    as_[i] = s; ad_[i] = d;
}

// ---- softmax pass 1: per-destination max (sortable-uint atomic max) -------
__global__ void k_edge_max(const long long* __restrict__ ei,
                           const float* __restrict__ as_, const float* __restrict__ ad_,
                           unsigned* __restrict__ mx, int E) {
    int e = blockIdx.x * blockDim.x + threadIdx.x;
    if (e >= E) return;
    int s = (int)ei[e], d = (int)ei[(size_t)E + e];
    float4 fs = ((const float4*)as_)[s];
    float4 fd = ((const float4*)ad_)[d];
    atomicMax(&mx[d * 4 + 0], fenc(lrelu(fs.x + fd.x)));
    atomicMax(&mx[d * 4 + 1], fenc(lrelu(fs.y + fd.y)));
    atomicMax(&mx[d * 4 + 2], fenc(lrelu(fs.z + fd.z)));
    atomicMax(&mx[d * 4 + 3], fenc(lrelu(fs.w + fd.w)));
}
__global__ void k_self_max(const float* __restrict__ as_, const float* __restrict__ ad_,
                           unsigned* __restrict__ mx, int N) {
    int n = blockIdx.x * blockDim.x + threadIdx.x;
    if (n >= N) return;
    float4 fs = ((const float4*)as_)[n];
    float4 fd = ((const float4*)ad_)[n];
    atomicMax(&mx[n * 4 + 0], fenc(lrelu(fs.x + fd.x)));
    atomicMax(&mx[n * 4 + 1], fenc(lrelu(fs.y + fd.y)));
    atomicMax(&mx[n * 4 + 2], fenc(lrelu(fs.z + fd.z)));
    atomicMax(&mx[n * 4 + 3], fenc(lrelu(fs.w + fd.w)));
}

// ---- softmax pass 2: per-destination sum of exp ---------------------------
__global__ void k_edge_sum(const long long* __restrict__ ei,
                           const float* __restrict__ as_, const float* __restrict__ ad_,
                           const unsigned* __restrict__ mx, float* __restrict__ den, int E) {
    int e = blockIdx.x * blockDim.x + threadIdx.x;
    if (e >= E) return;
    int s = (int)ei[e], d = (int)ei[(size_t)E + e];
    float4 fs = ((const float4*)as_)[s];
    float4 fd = ((const float4*)ad_)[d];
    uint4  mm = ((const uint4*)mx)[d];
    atomicAdd(&den[d * 4 + 0], __expf(lrelu(fs.x + fd.x) - fdec(mm.x)));
    atomicAdd(&den[d * 4 + 1], __expf(lrelu(fs.y + fd.y) - fdec(mm.y)));
    atomicAdd(&den[d * 4 + 2], __expf(lrelu(fs.z + fd.z) - fdec(mm.z)));
    atomicAdd(&den[d * 4 + 3], __expf(lrelu(fs.w + fd.w) - fdec(mm.w)));
}
__global__ void k_self_sum(const float* __restrict__ as_, const float* __restrict__ ad_,
                           const unsigned* __restrict__ mx, float* __restrict__ den, int N) {
    int n = blockIdx.x * blockDim.x + threadIdx.x;
    if (n >= N) return;
    float4 fs = ((const float4*)as_)[n];
    float4 fd = ((const float4*)ad_)[n];
    uint4  mm = ((const uint4*)mx)[n];
    atomicAdd(&den[n * 4 + 0], __expf(lrelu(fs.x + fd.x) - fdec(mm.x)));
    atomicAdd(&den[n * 4 + 1], __expf(lrelu(fs.y + fd.y) - fdec(mm.y)));
    atomicAdd(&den[n * 4 + 2], __expf(lrelu(fs.z + fd.z) - fdec(mm.z)));
    atomicAdd(&den[n * 4 + 3], __expf(lrelu(fs.w + fd.w) - fdec(mm.w)));
}

// ---- pass 3: out[d] += alpha * H[s] ; one wave per edge, 8 ch per lane ----
__global__ __launch_bounds__(256) void k_edge_accum(
    const long long* __restrict__ ei, const float* __restrict__ H,
    const float* __restrict__ as_, const float* __restrict__ ad_,
    const unsigned* __restrict__ mx, const float* __restrict__ den,
    float* __restrict__ out, int E) {
    int lane = threadIdx.x & 31;
    int e = blockIdx.x * 8 + (threadIdx.x >> 5);
    if (e >= E) return;
    int s = (int)ei[e], d = (int)ei[(size_t)E + e];
    int h = lane >> 3;  // 8 contiguous channels per lane -> one head per lane
    float v = lrelu(as_[s * HEADS + h] + ad_[d * HEADS + h]);
    float alpha = __expf(v - fdec(mx[d * HEADS + h])) / den[d * HEADS + h];
    const float* hs = H + (size_t)s * CH + lane * 8;
    float* op = out + (size_t)d * CH + lane * 8;
#pragma unroll
    for (int j = 0; j < 8; j++) atomicAdd(&op[j], alpha * hs[j]);
}
__global__ __launch_bounds__(256) void k_self_accum(
    const float* __restrict__ H, const float* __restrict__ as_,
    const float* __restrict__ ad_, const unsigned* __restrict__ mx,
    const float* __restrict__ den, float* __restrict__ out, int N) {
    int lane = threadIdx.x & 31;
    int n = blockIdx.x * 8 + (threadIdx.x >> 5);
    if (n >= N) return;
    int h = lane >> 3;
    float v = lrelu(as_[n * HEADS + h] + ad_[n * HEADS + h]);
    float alpha = __expf(v - fdec(mx[n * HEADS + h])) / den[n * HEADS + h];
    const float* hs = H + (size_t)n * CH + lane * 8;
    float* op = out + (size_t)n * CH + lane * 8;
#pragma unroll
    for (int j = 0; j < 8; j++) atomicAdd(&op[j], alpha * hs[j]);
}

// ---- fused bias + ReLU + eval-BatchNorm, dual f32/bf16 output -------------
__global__ void k_post(const float* __restrict__ acc, const float* __restrict__ b,
                       const float* __restrict__ g, const float* __restrict__ be,
                       const float* __restrict__ rm, const float* __restrict__ rv,
                       float* __restrict__ hf, __bf16* __restrict__ hb, size_t total) {
    size_t i = (size_t)blockIdx.x * blockDim.x + threadIdx.x;
    if (i >= total) return;
    int c = (int)(i & (CH - 1));
    float v = acc[i] + b[c];
    v = fmaxf(v, 0.0f);
    v = (v - rm[c]) * rsqrtf(rv[c] + BNEPS) * g[c] + be[c];
    hf[i] = v;
    hb[i] = (__bf16)v;
}

// ---- global mean pool -----------------------------------------------------
__global__ void k_pool(const float* __restrict__ h, const long long* __restrict__ batch,
                       float* __restrict__ pool, float* __restrict__ cnt, int N) {
    size_t i = (size_t)blockIdx.x * blockDim.x + threadIdx.x;
    if (i >= (size_t)N * CH) return;
    int n = (int)(i >> 8), c = (int)(i & 255);
    int b = (int)batch[n];
    atomicAdd(&pool[b * CH + c], h[i]);
    if (c == 0) atomicAdd(&cnt[b], 1.0f);
}
__global__ void k_pool_final(const float* __restrict__ pool, const float* __restrict__ cnt,
                             __bf16* __restrict__ pbf, int total) {
    int i = blockIdx.x * blockDim.x + threadIdx.x;
    if (i >= total) return;
    float cn = cnt[i >> 8];
    cn = cn < 1.0f ? 1.0f : cn;
    pbf[i] = (__bf16)(pool[i] / cn);
}

// ---- final 128x2 projection -----------------------------------------------
__global__ void k_final(const float* __restrict__ z, const float* __restrict__ lw2,
                        const float* __restrict__ lb2, float* __restrict__ out) {
    int i = blockIdx.x * blockDim.x + threadIdx.x;
    if (i >= NGR * NCLS) return;
    int g = i / NCLS, c = i % NCLS;
    float s = lb2[c];
    for (int k = 0; k < OUTD; k++) s += z[g * OUTD + k] * lw2[k * NCLS + c];
    out[i] = s;
}

// ---------------------------------------------------------------------------
#define CDIV(a, b) (unsigned)((((size_t)(a)) + (b) - 1) / (b))

extern "C" void kernel_launch(void* const* d_in, const int* in_sizes, int n_in,
                              void* d_out, int out_size, void* d_ws, size_t ws_size,
                              hipStream_t stream) {
    const float* x        = (const float*)d_in[0];
    const long long* ei   = (const long long*)d_in[1];
    const long long* batch= (const long long*)d_in[2];
    const float* W1  = (const float*)d_in[3];
    const float* as1 = (const float*)d_in[4];
    const float* ad1 = (const float*)d_in[5];
    const float* b1  = (const float*)d_in[6];
    const float* g1  = (const float*)d_in[7];
    const float* be1 = (const float*)d_in[8];
    const float* rm1 = (const float*)d_in[9];
    const float* rv1 = (const float*)d_in[10];
    const float* W2  = (const float*)d_in[11];
    const float* as2 = (const float*)d_in[12];
    const float* ad2 = (const float*)d_in[13];
    const float* b2  = (const float*)d_in[14];
    const float* g2  = (const float*)d_in[15];
    const float* be2 = (const float*)d_in[16];
    const float* rm2 = (const float*)d_in[17];
    const float* rv2 = (const float*)d_in[18];
    const float* lw1 = (const float*)d_in[19];
    const float* lb1 = (const float*)d_in[20];
    const float* lw2 = (const float*)d_in[21];
    const float* lb2 = (const float*)d_in[22];

    const int N = in_sizes[0] / F_IN;   // 50000
    const int E = in_sizes[1] / 2;      // 800000

    // workspace carve-out (~145 MB with reuse)
    char* ws = (char*)d_ws;
    size_t off = 0;
    auto alloc = [&](size_t bytes) -> char* {
        char* p = ws + off;
        off = (off + bytes + 255) & ~(size_t)255;
        return p;
    };
    __bf16* xbf  = (__bf16*)alloc((size_t)N * F_IN * 2);
    __bf16* w1t  = (__bf16*)alloc((size_t)CH * F_IN * 2);   // [256,128]
    __bf16* w2t  = (__bf16*)alloc((size_t)CH * CH * 2);     // [256,256]
    __bf16* l1t  = (__bf16*)alloc((size_t)OUTD * CH * 2);   // [128,256]
    float*  bufA = (float*)alloc((size_t)N * CH * 4);       // GEMM out / post f32
    float*  bufB = (float*)alloc((size_t)N * CH * 4);       // GAT accumulator
    __bf16* hbf  = (__bf16*)alloc((size_t)N * CH * 2);      // post bf16
    float*  asv  = (float*)alloc((size_t)N * HEADS * 4);
    float*  adv  = (float*)alloc((size_t)N * HEADS * 4);
    unsigned* mx = (unsigned*)alloc((size_t)N * HEADS * 4);
    float*  den  = (float*)alloc((size_t)N * HEADS * 4);
    float*  pool = (float*)alloc((size_t)NGR * CH * 4);
    float*  cnt  = (float*)alloc((size_t)NGR * 4);
    __bf16* pbf  = (__bf16*)alloc((size_t)NGR * CH * 2);
    float*  z    = (float*)alloc((size_t)NGR * OUTD * 4);

    // ---- prep: bf16 inputs, transposed bf16 weights ----
    k_cvt<<<CDIV((size_t)N * F_IN, 256), 256, 0, stream>>>(x, xbf, (size_t)N * F_IN);
    k_wt<<<CDIV(F_IN * CH, 256), 256, 0, stream>>>(W1, w1t, F_IN, CH);
    k_wt<<<CDIV(CH * CH, 256), 256, 0, stream>>>(W2, w2t, CH, CH);
    k_wt<<<CDIV(CH * OUTD, 256), 256, 0, stream>>>(lw1, l1t, CH, OUTD);

    auto gat_layer = [&](const __bf16* Ain, const __bf16* Wt, int K,
                         const float* attS, const float* attD, const float* bias,
                         const float* g, const float* be, const float* rm,
                         const float* rv) {
        unsigned tiles = CDIV(N, 32) * (CH / 32);
        k_gemm_bf16<false, false><<<CDIV(tiles, 8), 256, 0, stream>>>(
            Ain, Wt, nullptr, bufA, N, K, CH);
        k_att<<<CDIV(N * HEADS, 256), 256, 0, stream>>>(bufA, attS, attD, asv, adv, N);
        hipMemsetAsync(mx, 0, (size_t)N * HEADS * 4, stream);   // 0 < enc(-inf)
        hipMemsetAsync(den, 0, (size_t)N * HEADS * 4, stream);
        hipMemsetAsync(bufB, 0, (size_t)N * CH * 4, stream);
        k_edge_max<<<CDIV(E, 256), 256, 0, stream>>>(ei, asv, adv, mx, E);
        k_self_max<<<CDIV(N, 256), 256, 0, stream>>>(asv, adv, mx, N);
        k_edge_sum<<<CDIV(E, 256), 256, 0, stream>>>(ei, asv, adv, mx, den, E);
        k_self_sum<<<CDIV(N, 256), 256, 0, stream>>>(asv, adv, mx, den, N);
        k_edge_accum<<<CDIV(E, 8), 256, 0, stream>>>(ei, bufA, asv, adv, mx, den, bufB, E);
        k_self_accum<<<CDIV(N, 8), 256, 0, stream>>>(bufA, asv, adv, mx, den, bufB, N);
        k_post<<<CDIV((size_t)N * CH, 256), 256, 0, stream>>>(
            bufB, bias, g, be, rm, rv, bufA, hbf, (size_t)N * CH);
    };

    gat_layer(xbf, w1t, F_IN, as1, ad1, b1, g1, be1, rm1, rv1);   // layer 1
    gat_layer(hbf, w2t, CH,   as2, ad2, b2, g2, be2, rm2, rv2);   // layer 2

    // ---- mean pool + MLP head ----
    hipMemsetAsync(pool, 0, (size_t)NGR * CH * 4, stream);
    hipMemsetAsync(cnt, 0, (size_t)NGR * 4, stream);
    k_pool<<<CDIV((size_t)N * CH, 256), 256, 0, stream>>>(bufA, batch, pool, cnt, N);
    k_pool_final<<<CDIV(NGR * CH, 256), 256, 0, stream>>>(pool, cnt, pbf, NGR * CH);
    unsigned tiles3 = CDIV(NGR, 32) * (OUTD / 32);
    k_gemm_bf16<true, true><<<CDIV(tiles3, 8), 256, 0, stream>>>(
        pbf, l1t, lb1, z, NGR, CH, OUTD);
    k_final<<<CDIV(NGR * NCLS, 256), 256, 0, stream>>>(z, lw2, lb2, (float*)d_out);
}
